// TemporalAttentionTVModel_16114717294632
// MI455X (gfx1250) — matrix operand order repaired and checked by reading, not compiled, not run on hardware
//
#include <hip/hip_runtime.h>

// ---------------------------------------------------------------------------
// Relative-position multi-head attention + residual + LayerNorm for MI455X
// (gfx1250, wave32, WMMA). bf16 A/B with f32 accumulation everywhere.
// The whole attention core (Qrel stripe, scores, softmax, attn@V + T@relV)
// is fused into ONE kernel that keeps all intermediates in LDS:
//   phase 0: Qrel stripe [16,1088] = Q_stripe @ relK^T          (LDS, f32)
//   phase 1: scores stripe [16,1024] = QK^T/8 + gather(Qrel)    (LDS, f32)
//   phase 2: softmax; emit normalized attn [16,1024] and banded T [16,1088]
//            as bf16 into LDS (aliasing the dead Qrel stripe)
//   phase 3: ctx stripe [16,64] = attn @ V + T @ relV  (split-K over waves,
//            LDS reduction), stored bf16 head-interleaved.
// Nothing but Q/K/V^T and ctx ever touches HBM for the attention core.
// ---------------------------------------------------------------------------

#define S_LEN 1024
#define DMODEL 512
#define NHEAD 8
#define DK 64
#define BATCH 4
#define BH (BATCH * NHEAD)          // 32
#define PMAX 512
#define RELN 1025                   // 2P+1
#define RELP 1088                   // padded rel length (68*16, 34*32)

typedef __attribute__((ext_vector_type(16))) __bf16 bf16x16;
typedef __attribute__((ext_vector_type(8)))  float  f32x8;

union Frag { bf16x16 v; uint4 u[2]; };

__device__ __forceinline__ __bf16 f2bf(float f) {
    unsigned u = __builtin_bit_cast(unsigned, f);
    unsigned r = (u + 0x7FFFu + ((u >> 16) & 1u)) >> 16;     // RNE
    unsigned short s = (unsigned short)r;
    return __builtin_bit_cast(__bf16, s);
}

__device__ __forceinline__ f32x8 wmma_bf16(const Frag& a, const Frag& b, f32x8 c) {
    return __builtin_amdgcn_wmma_f32_16x16x32_bf16(false, a.v, false, b.v,
                                                   (short)0, c, false, false);
}

// ---------------------------------------------------------------------------
// Generic batched GEMM: C = scale * (A @ Bt^T) (+ epilogue per MODE)
//   A : [M,K] bf16 row-major (lda), batched by strideAz
//   Bt: [N,K] bf16 row-major (ldb), batched by strideBz  (i.e. B is K x N)
// Block = 256 threads = 8 waves; wave tile 32x32 (2x2 WMMA 16x16x32 bf16).
// Block tile 128(M) x 64(N). All launch dims are exact multiples -> no guards.
// MODE 0: f32 store (+optional bias[n])
// MODE 2: bf16 store, head-split [B,H,S,DK]   (from m = b*S+s, n = h*DK+d)
// MODE 3: bf16 store, head-split transposed [B,H,DK,S]
// ---------------------------------------------------------------------------
template<int MODE>
__global__ __launch_bounds__(256) void gemm_bt(
    const __bf16* __restrict__ A,  int lda, long long strideAz,
    const __bf16* __restrict__ Bt, int ldb, long long strideBz,
    void* __restrict__ Cv, int ldc,
    int M, int N, int K, float scale,
    const float* __restrict__ bias)
{
    const int z = blockIdx.z;
    A  += (long long)z * strideAz;
    Bt += (long long)z * strideBz;

    const int lane  = threadIdx.x & 31;
    const int wid   = threadIdx.x >> 5;
    const int waveN = wid & 1;
    const int waveM = wid >> 1;
    const int m0 = blockIdx.y * 128 + waveM * 32;
    const int n0 = blockIdx.x * 64  + waveN * 32;
    const int half = lane >> 4;
    const int l15  = lane & 15;

    f32x8 acc[2][2] = {};

    for (int k0 = 0; k0 < K; k0 += 32) {
        Frag a[2], b[2];
#pragma unroll
        for (int t = 0; t < 2; ++t) {
            // A fragment 16x32: lane l holds row (l&15); K = k0 + 8*half + {0..7, 16..23}
            const __bf16* ap = A + (size_t)(m0 + t * 16 + l15) * lda + k0 + half * 8;
            a[t].u[0] = *(const uint4*)(ap);
            a[t].u[1] = *(const uint4*)(ap + 16);
            // B fragment 32x16 (KxN): lane l holds col N=(l&15); K = k0 + 8*half + {0..7, 16..23}
            const __bf16* bp = Bt + (size_t)(n0 + t * 16 + l15) * ldb + k0 + half * 8;
            b[t].u[0] = *(const uint4*)(bp);
            b[t].u[1] = *(const uint4*)(bp + 16);
        }
        if (k0 + 32 < K) {   // hint next K-tile into cache (global_prefetch_b8)
            __builtin_prefetch(A  + (size_t)(m0 + l15) * lda + k0 + 32, 0, 1);
            __builtin_prefetch(Bt + (size_t)(n0 + l15) * ldb + k0 + 32, 0, 1);
        }
#pragma unroll
        for (int mt = 0; mt < 2; ++mt)
#pragma unroll
            for (int nt = 0; nt < 2; ++nt)
                acc[mt][nt] = wmma_bf16(a[mt], b[nt], acc[mt][nt]);
    }

    // Epilogue. C/D layout: VGPR g -> local m = g + 8*half, local n = lane&15.
#pragma unroll
    for (int mt = 0; mt < 2; ++mt)
#pragma unroll
    for (int nt = 0; nt < 2; ++nt)
#pragma unroll
    for (int g = 0; g < 8; ++g) {
        const int m = m0 + mt * 16 + g + half * 8;
        const int n = n0 + nt * 16 + l15;
        float v = acc[mt][nt][g] * scale;
        if constexpr (MODE == 0) {
            if (bias) v += bias[n];
            ((float*)Cv)[(size_t)m * ldc + n] = v;
        } else if constexpr (MODE == 2) {           // Q/K: [B,H,S,DK]
            const int bb = m >> 10, ss = m & 1023, hh = n >> 6, dd = n & 63;
            ((__bf16*)Cv)[((((size_t)bb * NHEAD + hh) * S_LEN + ss) << 6) + dd] = f2bf(v);
        } else if constexpr (MODE == 3) {           // V^T: [B,H,DK,S]
            const int bb = m >> 10, ss = m & 1023, hh = n >> 6, dd = n & 63;
            ((__bf16*)Cv)[((((size_t)bb * NHEAD + hh) * DK + dd) << 10) + ss] = f2bf(v);
        }
    }
}

// ---------------------------------------------------------------------------
// Fully fused relative-position attention core. One block = 16 query rows of
// one (b,h) head. 8 waves, 256 threads. LDS = 64KB + 69.6KB.
// ---------------------------------------------------------------------------
__global__ __launch_bounds__(256) void attention_fused_kernel(
    const __bf16* __restrict__ Q,      // [BH, S, DK]
    const __bf16* __restrict__ Kb,     // [BH, S, DK]
    const __bf16* __restrict__ VT,     // [BH, DK, S]
    const __bf16* __restrict__ relKp,  // [RELP, DK]  (zero-padded rows)
    const __bf16* __restrict__ relVT,  // [DK, RELP]  (zero-padded cols)
    __bf16* __restrict__ ctx)          // [B*S, DMODEL] head-interleaved
{
    __shared__ float sc[16 * S_LEN];   // scores stripe; later ctx partials
    __shared__ float qs[16 * RELP];    // Qrel stripe; later attn/T bf16 (alias)
    __bf16* attn_lds = (__bf16*)qs;            // 16 x 1024 bf16 (32 KB)
    __bf16* t_lds    = attn_lds + 16 * S_LEN;  // 16 x 1088 bf16 (34.8 KB)

    const int z    = blockIdx.y;       // (b*H + h)
    const int m0   = blockIdx.x * 16;  // query row-block within head
    const int lane = threadIdx.x & 31;
    const int w    = threadIdx.x >> 5;
    const int half = lane >> 4;
    const int l15  = lane & 15;

    const __bf16* Qh  = Q  + (size_t)z * S_LEN * DK;
    const __bf16* Kh  = Kb + (size_t)z * S_LEN * DK;
    const __bf16* VTh = VT + (size_t)z * DK * S_LEN;

    // Q fragments for this block's 16 rows, K = 0..63 (reused phases 0 & 1)
    Frag a0, a1;
    {
        const __bf16* ap = Qh + (size_t)(m0 + l15) * DK + half * 8;
        a0.u[0] = *(const uint4*)(ap);       a0.u[1] = *(const uint4*)(ap + 16);
        a1.u[0] = *(const uint4*)(ap + 32);  a1.u[1] = *(const uint4*)(ap + 48);
    }

    // ---- phase 0: Qrel stripe qs[16][RELP] = Q_stripe @ relK^T ------------
    for (int t = w; t < RELP / 16; t += 8) {
        const int n0 = t * 16;
        const __bf16* bp = relKp + (size_t)(n0 + l15) * DK + half * 8;
        Frag b0, b1;
        b0.u[0] = *(const uint4*)(bp);       b0.u[1] = *(const uint4*)(bp + 16);
        b1.u[0] = *(const uint4*)(bp + 32);  b1.u[1] = *(const uint4*)(bp + 48);
        f32x8 acc = {};
        acc = wmma_bf16(a0, b0, acc);
        acc = wmma_bf16(a1, b1, acc);
#pragma unroll
        for (int g = 0; g < 8; ++g)
            qs[(g + half * 8) * RELP + n0 + l15] = acc[g];
    }
    __syncthreads();

    // ---- phase 1: scores stripe sc[16][S] = QK^T/8 + gather(Qrel) ---------
    for (int t = 0; t < 8; ++t) {
        const int n0 = (w * 8 + t) * 16;
        const __bf16* bp = Kh + (size_t)(n0 + l15) * DK + half * 8;
        Frag b0, b1;
        b0.u[0] = *(const uint4*)(bp);       b0.u[1] = *(const uint4*)(bp + 16);
        b1.u[0] = *(const uint4*)(bp + 32);  b1.u[1] = *(const uint4*)(bp + 48);
        f32x8 acc = {};
        acc = wmma_bf16(a0, b0, acc);
        acc = wmma_bf16(a1, b1, acc);
#pragma unroll
        for (int g = 0; g < 8; ++g) {
            const int lm = g + half * 8;          // local row 0..15
            const int n  = n0 + l15;              // column j
            const int i  = m0 + lm;               // global query row
            int rel = n - i;
            rel = rel < -PMAX ? -PMAX : (rel > PMAX ? PMAX : rel);
            sc[lm * S_LEN + n] = acc[g] * 0.125f + qs[lm * RELP + rel + PMAX];
        }
    }
    __syncthreads();

    // ---- phase 2: softmax rows {2w, 2w+1}; emit attn/T bf16 into LDS ------
    for (int rr = 0; rr < 2; ++rr) {
        const int lr = 2 * w + rr;
        const int i  = m0 + lr;
        float* srow = sc + lr * S_LEN;

        float mx = -3.0e38f;
        for (int j = lane; j < S_LEN; j += 32) mx = fmaxf(mx, srow[j]);
#pragma unroll
        for (int o = 16; o > 0; o >>= 1) mx = fmaxf(mx, __shfl_xor(mx, o));

        float sum = 0.0f, s0 = 0.0f, s1 = 0.0f;
        for (int j = lane; j < S_LEN; j += 32) {
            float e = __expf(srow[j] - mx);
            srow[j] = e;
            sum += e;
            if (j <= i - PMAX) s0 += e;     // clamp-low bucket  -> T[i,0]
            if (j >= i + PMAX) s1 += e;     // clamp-high bucket -> T[i,2P]
        }
#pragma unroll
        for (int o = 16; o > 0; o >>= 1) {
            sum += __shfl_xor(sum, o);
            s0  += __shfl_xor(s0, o);
            s1  += __shfl_xor(s1, o);
        }
        const float inv = 1.0f / sum;

        for (int j = lane; j < S_LEN; j += 32)
            attn_lds[lr * S_LEN + j] = f2bf(srow[j] * inv);

        for (int r = lane; r < RELP; r += 32) {
            float v;
            if (r == 0)                 v = s0 * inv;
            else if (r == 2 * PMAX)     v = s1 * inv;
            else if (r < 2 * PMAX) {
                const int j = i + r - PMAX;
                v = (j >= 0 && j < S_LEN) ? srow[j] * inv : 0.0f;
            } else                      v = 0.0f;    // padding r > 2P
            t_lds[lr * RELP + r] = f2bf(v);
        }
    }
    __syncthreads();

    // ---- phase 3: ctx stripe = attn @ V + T @ relV ------------------------
    // 8 waves = 4 column tiles x 2 K-partitions (attn part | T part)
    {
        const int nt = w & 3;                    // 16-col tile of DK=64
        f32x8 acc = {};
        if ((w >> 2) == 0) {
            for (int ks = 0; ks < S_LEN; ks += 32) {          // attn @ V
                Frag af, fb;
                const __bf16* ar = attn_lds + l15 * S_LEN + ks + half * 8;
                af.u[0] = *(const uint4*)(ar);
                af.u[1] = *(const uint4*)(ar + 16);
                const __bf16* bp = VTh + (size_t)(nt * 16 + l15) * S_LEN + ks + half * 8;
                fb.u[0] = *(const uint4*)(bp);
                fb.u[1] = *(const uint4*)(bp + 16);
                acc = wmma_bf16(af, fb, acc);
            }
        } else {
            for (int ks = 0; ks < RELP; ks += 32) {           // T @ relV
                Frag af, fb;
                const __bf16* ar = t_lds + l15 * RELP + ks + half * 8;
                af.u[0] = *(const uint4*)(ar);
                af.u[1] = *(const uint4*)(ar + 16);
                const __bf16* bp = relVT + (size_t)(nt * 16 + l15) * RELP + ks + half * 8;
                fb.u[0] = *(const uint4*)(bp);
                fb.u[1] = *(const uint4*)(bp + 16);
                acc = wmma_bf16(af, fb, acc);
            }
        }
#pragma unroll
        for (int g = 0; g < 8; ++g)                 // partial tile -> LDS
            sc[w * 256 + (g + half * 8) * 16 + l15] = acc[g];
    }
    __syncthreads();

    // ---- reduce the two K-partitions, store ctx bf16 ----------------------
    {
        const int bb = z >> 3, hh = z & 7;
        for (int e = threadIdx.x; e < 16 * DK; e += 256) {
            const int lm = e >> 6, c = e & 63;
            const int nt = c >> 4, cc = c & 15;
            float v = sc[nt * 256 + lm * 16 + cc]
                    + sc[(nt + 4) * 256 + lm * 16 + cc];
            ctx[((size_t)(bb * S_LEN + m0 + lm)) * DMODEL + hh * DK + c] = f2bf(v);
        }
    }
}

// ---------------------------------------------------------------------------
// Elementwise converters
// ---------------------------------------------------------------------------
__global__ __launch_bounds__(256) void cvt_f32_bf16(const float* __restrict__ s,
                                                    __bf16* __restrict__ d, int n) {
    int i = blockIdx.x * 256 + threadIdx.x;
    if (i < n) d[i] = f2bf(s[i]);
}

__global__ __launch_bounds__(256) void relk_pad(const float* __restrict__ relK,
                                                __bf16* __restrict__ d) {
    int i = blockIdx.x * 256 + threadIdx.x;          // over RELP*DK
    if (i < RELP * DK) {
        int r = i >> 6, c = i & 63;
        d[i] = (r < RELN) ? f2bf(relK[r * DK + c]) : f2bf(0.0f);
    }
}

__global__ __launch_bounds__(256) void relv_tr(const float* __restrict__ relV,
                                               __bf16* __restrict__ d) {
    int i = blockIdx.x * 256 + threadIdx.x;          // over DK*RELP
    if (i < DK * RELP) {
        int dd = i / RELP, r = i % RELP;
        d[i] = (r < RELN) ? f2bf(relV[r * DK + dd]) : f2bf(0.0f);
    }
}

// ---------------------------------------------------------------------------
// LayerNorm(out + x) over D=512
// ---------------------------------------------------------------------------
__global__ __launch_bounds__(256) void ln_kernel(const float* __restrict__ o,
                                                 const float* __restrict__ x,
                                                 const float* __restrict__ gamma,
                                                 const float* __restrict__ beta,
                                                 float* __restrict__ y) {
    __shared__ float red[256];
    const size_t row = blockIdx.x;
    const float* orow = o + row * DMODEL;
    const float* xrow = x + row * DMODEL;
    const int t = threadIdx.x;
    float v0 = orow[t] + xrow[t];
    float v1 = orow[t + 256] + xrow[t + 256];
    red[t] = v0 + v1; __syncthreads();
    for (int s = 128; s > 0; s >>= 1) { if (t < s) red[t] += red[t + s]; __syncthreads(); }
    const float mu = red[0] * (1.0f / DMODEL); __syncthreads();
    const float d0 = v0 - mu, d1 = v1 - mu;
    red[t] = d0 * d0 + d1 * d1; __syncthreads();
    for (int s = 128; s > 0; s >>= 1) { if (t < s) red[t] += red[t + s]; __syncthreads(); }
    const float inv = rsqrtf(red[0] * (1.0f / DMODEL) + 1e-5f);
    y[row * DMODEL + t]       = d0 * inv * gamma[t] + beta[t];
    y[row * DMODEL + t + 256] = d1 * inv * gamma[t + 256] + beta[t + 256];
}

// ---------------------------------------------------------------------------
// Host orchestration
// ---------------------------------------------------------------------------
extern "C" void kernel_launch(void* const* d_in, const int* in_sizes, int n_in,
                              void* d_out, int out_size, void* d_ws, size_t ws_size,
                              hipStream_t stream) {
    (void)in_sizes; (void)n_in; (void)out_size; (void)ws_size;
    const float* x     = (const float*)d_in[0];
    const float* Wq    = (const float*)d_in[1];
    const float* Wk    = (const float*)d_in[2];
    const float* Wv    = (const float*)d_in[3];
    const float* Wo    = (const float*)d_in[4];
    const float* bo    = (const float*)d_in[5];
    const float* relK  = (const float*)d_in[6];
    const float* relV  = (const float*)d_in[7];
    const float* gamma = (const float*)d_in[8];
    const float* beta  = (const float*)d_in[9];

    const int NX = BATCH * S_LEN * DMODEL;          // 2,097,152
    const int NW = DMODEL * DMODEL;                 // 262,144

    char* p = (char*)d_ws;
    auto take = [&](size_t bytes) -> char* {
        char* r = p; p += (bytes + 255) & ~(size_t)255; return r;
    };
    __bf16* x_bf    = (__bf16*)take((size_t)NX * 2);
    __bf16* Wq_bf   = (__bf16*)take((size_t)NW * 2);
    __bf16* Wk_bf   = (__bf16*)take((size_t)NW * 2);
    __bf16* Wv_bf   = (__bf16*)take((size_t)NW * 2);
    __bf16* Wo_bf   = (__bf16*)take((size_t)NW * 2);
    __bf16* relK_bf = (__bf16*)take((size_t)RELP * DK * 2);
    __bf16* relVT   = (__bf16*)take((size_t)DK * RELP * 2);
    __bf16* Q_bf    = (__bf16*)take((size_t)NX * 2);
    __bf16* K_bf    = (__bf16*)take((size_t)NX * 2);
    __bf16* VT_bf   = (__bf16*)take((size_t)NX * 2);
    __bf16* ctx_bf  = (__bf16*)take((size_t)NX * 2);
    float*  out_f32 = (float*) take((size_t)NX * 4);

    // --- conversions / layout prep -----------------------------------------
    cvt_f32_bf16<<<(NX + 255) / 256, 256, 0, stream>>>(x,  x_bf,  NX);
    cvt_f32_bf16<<<(NW + 255) / 256, 256, 0, stream>>>(Wq, Wq_bf, NW);
    cvt_f32_bf16<<<(NW + 255) / 256, 256, 0, stream>>>(Wk, Wk_bf, NW);
    cvt_f32_bf16<<<(NW + 255) / 256, 256, 0, stream>>>(Wv, Wv_bf, NW);
    cvt_f32_bf16<<<(NW + 255) / 256, 256, 0, stream>>>(Wo, Wo_bf, NW);
    relk_pad<<<(RELP * DK + 255) / 256, 256, 0, stream>>>(relK, relK_bf);
    relv_tr <<<(DK * RELP + 255) / 256, 256, 0, stream>>>(relV, relVT);

    // --- projections: Q,K head-split; V head-split transposed --------------
    gemm_bt<2><<<dim3(DMODEL / 64, (BATCH * S_LEN) / 128, 1), 256, 0, stream>>>(
        x_bf, DMODEL, 0, Wq_bf, DMODEL, 0, Q_bf, 0,
        BATCH * S_LEN, DMODEL, DMODEL, 1.0f, nullptr);
    gemm_bt<2><<<dim3(DMODEL / 64, (BATCH * S_LEN) / 128, 1), 256, 0, stream>>>(
        x_bf, DMODEL, 0, Wk_bf, DMODEL, 0, K_bf, 0,
        BATCH * S_LEN, DMODEL, DMODEL, 1.0f, nullptr);
    gemm_bt<3><<<dim3(DMODEL / 64, (BATCH * S_LEN) / 128, 1), 256, 0, stream>>>(
        x_bf, DMODEL, 0, Wv_bf, DMODEL, 0, VT_bf, 0,
        BATCH * S_LEN, DMODEL, DMODEL, 1.0f, nullptr);

    // --- fully fused attention core ----------------------------------------
    attention_fused_kernel<<<dim3(S_LEN / 16, BH), 256, 0, stream>>>(
        Q_bf, K_bf, VT_bf, relK_bf, relVT, ctx_bf);

    // --- out = ctx @ Wo^T + bo  (f32) --------------------------------------
    gemm_bt<0><<<dim3(DMODEL / 64, (BATCH * S_LEN) / 128, 1), 256, 0, stream>>>(
        ctx_bf, DMODEL, 0, Wo_bf, DMODEL, 0, out_f32, DMODEL,
        BATCH * S_LEN, DMODEL, DMODEL, 1.0f, bo);

    // --- LayerNorm(out + x) -> d_out ---------------------------------------
    ln_kernel<<<BATCH * S_LEN, 256, 0, stream>>>(out_f32, x, gamma, beta,
                                                 (float*)d_out);
}
